// MLPScorer_56581899158180
// MI455X (gfx1250) — matrix operand order repaired and checked by reading, not compiled, not run on hardware
//
#include <hip/hip_runtime.h>

// ---------------------------------------------------------------------------
// Problem constants (match reference)
// ---------------------------------------------------------------------------
#define NN   2048
#define MM   2048
#define DIMK 128          // hidden dim (= K of the reduction)
#define TILE 128          // output tile per block in kernel 2
#define D2   (DIMK / 2)   // d-dimension in half2 units
#define PAD2 1            // LDS padding in h2 units (row stride 65 dwords)

typedef __attribute__((ext_vector_type(16))) __bf16    v16bf;
typedef __attribute__((ext_vector_type(8)))  float     v8f;
typedef __attribute__((ext_vector_type(2)))  _Float16  h2;

// ---------------------------------------------------------------------------
// Kernel 2 (defined first so the hot loop shows in the disasm snippet):
// out[i,j] = sum_d relu(H1[i,d] + H2[j,d]) * w2[d] + b2
// 256 threads/block, 128x128 tile, 8x8 register micro-tile/thread.
// Packed-f16 inner loop: v_pk_add_f16 + v_pk_max_num_f16 + v_pk_fma_f16
// process 2 d-values per instruction -> 1.5 VALU instr / output-element-d.
// ---------------------------------------------------------------------------
__global__ __launch_bounds__(256) void mlp_score_kernel(
    const _Float16* __restrict__ H1, const _Float16* __restrict__ H2,
    const float* __restrict__ W2, const float* __restrict__ B2,
    float* __restrict__ out)
{
    __shared__ h2 As[TILE][D2 + PAD2];   // 128 x 65 dwords = 33.3 KB
    __shared__ h2 Bs[TILE][D2 + PAD2];
    __shared__ h2 w2s[D2];

    const int tid = threadIdx.x;
    const int i0  = blockIdx.y * TILE;
    const int j0  = blockIdx.x * TILE;

    // Cooperative tile load: 16384 halves per tile, 8 x (8 halves) per thread,
    // 16B-aligned uint4 global loads, conflict-free LDS stores.
    #pragma unroll
    for (int c = 0; c < 8; ++c) {
        const int flat = c * 2048 + tid * 8;   // half-index within tile
        const int row  = flat >> 7;            // /128
        const int d    = flat & 127;           // multiple of 8
        const int d2   = d >> 1;               // h2 index, multiple of 4
        const uint4 a  = *(const uint4*)(H1 + (size_t)(i0 + row) * DIMK + d);
        const uint4 b  = *(const uint4*)(H2 + (size_t)(j0 + row) * DIMK + d);
        As[row][d2 + 0] = __builtin_bit_cast(h2, a.x);
        As[row][d2 + 1] = __builtin_bit_cast(h2, a.y);
        As[row][d2 + 2] = __builtin_bit_cast(h2, a.z);
        As[row][d2 + 3] = __builtin_bit_cast(h2, a.w);
        Bs[row][d2 + 0] = __builtin_bit_cast(h2, b.x);
        Bs[row][d2 + 1] = __builtin_bit_cast(h2, b.y);
        Bs[row][d2 + 2] = __builtin_bit_cast(h2, b.z);
        Bs[row][d2 + 3] = __builtin_bit_cast(h2, b.w);
    }
    if (tid < D2) {
        h2 w;
        w[0] = (_Float16)W2[2 * tid + 0];
        w[1] = (_Float16)W2[2 * tid + 1];
        w2s[tid] = w;
    }
    __syncthreads();

    const int tx = tid & 15;
    const int ty = tid >> 4;
    const int ri = ty * 8;                 // 8 i-rows for this thread
    const int rj = tx * 8;                 // 8 j-cols for this thread

    h2 acc[8][8];                          // even/odd-d partial sums
    const h2 hz = {};
    #pragma unroll
    for (int p = 0; p < 8; ++p)
        #pragma unroll
        for (int q = 0; q < 8; ++q) acc[p][q] = hz;

    #pragma unroll 4
    for (int d2 = 0; d2 < D2; ++d2) {      // each step covers 2 d-values
        const h2 w = w2s[d2];
        h2 av[8], bv[8];
        #pragma unroll
        for (int p = 0; p < 8; ++p) av[p] = As[ri + p][d2];  // broadcast/half-wave
        #pragma unroll
        for (int q = 0; q < 8; ++q) bv[q] = Bs[rj + q][d2];  // <=2-way conflict
        #pragma unroll
        for (int p = 0; p < 8; ++p) {
            #pragma unroll
            for (int q = 0; q < 8; ++q) {
                h2 t = av[p] + bv[q];                       // v_pk_add_f16
                t = __builtin_elementwise_max(t, hz);       // v_pk_max_num_f16
                acc[p][q] = __builtin_elementwise_fma(t, w, acc[p][q]); // v_pk_fma_f16
            }
        }
    }

    const float bias2 = B2[0];
    #pragma unroll
    for (int p = 0; p < 8; ++p) {
        float* orow = out + (size_t)(i0 + ri + p) * MM + j0 + rj;
        #pragma unroll
        for (int q = 0; q < 8; q += 4) {
            float4 v;
            v.x = (float)acc[p][q + 0][0] + (float)acc[p][q + 0][1] + bias2;
            v.y = (float)acc[p][q + 1][0] + (float)acc[p][q + 1][1] + bias2;
            v.z = (float)acc[p][q + 2][0] + (float)acc[p][q + 2][1] + bias2;
            v.w = (float)acc[p][q + 3][0] + (float)acc[p][q + 3][1] + bias2;
            *(float4*)(orow + q) = v;
        }
    }
}

// ---------------------------------------------------------------------------
// Kernel 1: WMMA projections -> f16 intermediates.
//   sel==0 : H1[m,d] = f16( sum_k z_i[m,k]*W1[k,d] + b1[d] )
//   sel==1 : H2[m,d] = f16( sum_k z_j[m,k]*W1[128+k,d] )
// One wave per 16-row stripe; 8 N-tiles x 4 K-steps = 32 WMMAs.
// ---------------------------------------------------------------------------
__global__ __launch_bounds__(32) void hproj_wmma(
    const float* __restrict__ Zi, const float* __restrict__ Zj,
    const float* __restrict__ W1, const float* __restrict__ b1,
    _Float16* __restrict__ H1, _Float16* __restrict__ H2)
{
    const int bi   = blockIdx.x;            // 0..255
    const int sel  = bi >> 7;               // 0 -> H1, 1 -> H2
    const int m0   = (bi & 127) * 16;
    const int lane = threadIdx.x;           // wave32

    const float* __restrict__ src  = sel ? Zj : Zi;
    _Float16*    __restrict__ Hout = sel ? H2 : H1;
    const int kOff = sel * DIMK;            // which half of W1

    const int mrow  = lane & 15;            // A: row within tile
    const int khalf = (lane >> 4) * 16;     // A: K sub-offset per half-wave
    const int ncol  = lane & 15;            // B/D: column within tile
    const int kb    = (lane >> 4) * 16;     // B: K offset per half-wave

    v8f acc[8];
    #pragma unroll
    for (int nt = 0; nt < 8; ++nt) acc[nt] = (v8f){};

    #pragma unroll
    for (int ks = 0; ks < 4; ++ks) {
        const int k0 = ks * 32;

        // A tile: 16x32 bf16. Lane l (l<16): row=l, K = {khalf..khalf+15}
        const float* arow = src + (size_t)(m0 + mrow) * DIMK + k0 + khalf;
        v16bf a;
        #pragma unroll
        for (int e = 0; e < 8; ++e) a[e]     = (__bf16)arow[e];
        #pragma unroll
        for (int e = 0; e < 8; ++e) a[8 + e] = (__bf16)arow[8 + e];

        #pragma unroll
        for (int nt = 0; nt < 8; ++nt) {
            const int d0 = nt * 16;
            // B tile: 32x16 bf16. Lanes 0-15 hold K=0..15, lanes 16-31 K=16..31.
            const float* bcol = W1 + (size_t)(kOff + k0 + kb) * DIMK + d0 + ncol;
            v16bf b;
            #pragma unroll
            for (int e = 0; e < 16; ++e) b[e] = (__bf16)bcol[(size_t)e * DIMK];

            acc[nt] = __builtin_amdgcn_wmma_f32_16x16x32_bf16(
                /*neg_a=*/false, a, /*neg_b=*/false, b,
                /*c_mod=*/(short)0, acc[nt],
                /*reuse_a=*/false, /*reuse_b=*/false);
        }
    }

    // D layout: VGPR v -> M = v (+8 for lanes 16-31), N = lane&15.
    const int rbase = m0 + ((lane >> 4) ? 8 : 0);
    #pragma unroll
    for (int nt = 0; nt < 8; ++nt) {
        const int col = nt * 16 + ncol;
        const float add = sel ? 0.0f : b1[col];   // fold b1 into H1
        #pragma unroll
        for (int v = 0; v < 8; ++v) {
            Hout[(size_t)(rbase + v) * DIMK + col] = (_Float16)(acc[nt][v] + add);
        }
    }
}

// ---------------------------------------------------------------------------
// Launch
// ---------------------------------------------------------------------------
extern "C" void kernel_launch(void* const* d_in, const int* in_sizes, int n_in,
                              void* d_out, int out_size, void* d_ws, size_t ws_size,
                              hipStream_t stream) {
    const float* z_i = (const float*)d_in[0];   // [2048,128]
    const float* z_j = (const float*)d_in[1];   // [2048,128]
    const float* W1  = (const float*)d_in[2];   // [256,128]
    const float* b1  = (const float*)d_in[3];   // [128]
    const float* W2  = (const float*)d_in[4];   // [128,1]
    const float* b2  = (const float*)d_in[5];   // [1]
    float* out = (float*)d_out;                 // [2048,2048]

    _Float16* H1 = (_Float16*)d_ws;             // [2048,128]  (b1 folded in)
    _Float16* H2 = H1 + (size_t)NN * DIMK;      // [2048,128]

    hproj_wmma<<<dim3(256), dim3(32), 0, stream>>>(z_i, z_j, W1, b1, H1, H2);
    mlp_score_kernel<<<dim3(MM / TILE, NN / TILE), dim3(256), 0, stream>>>(
        H1, H2, W2, b2, out);
}
